// RNN_22608707846504
// MI455X (gfx1250) — matrix operand order repaired and checked by reading, not compiled, not run on hardware
//
#include <hip/hip_runtime.h>

typedef __attribute__((ext_vector_type(16))) _Float16 v16h;
typedef __attribute__((ext_vector_type(8)))  _Float16 v8h;
typedef __attribute__((ext_vector_type(8)))  float    v8f;
typedef __attribute__((ext_vector_type(4)))  float    v4f;

#define B_   64
#define T_   2048
#define DIN  64
#define H_   256
#define DOUT 64

#define HPAD 8
#define HS (H_ + HPAD)   // 264 halves/row -> 528 B (16B-aligned stride, bank-spread)
#define XS (DIN + 8)     // 72 halves/row  -> 144 B

// B-fragment K index for half j, lane-group g (16-bit B 32x16)
__device__ __forceinline__ int kB(int j, int g) {
    return (j < 8) ? (g * 8 + j) : (16 + g * 8 + (j - 8));
}

// ---------------------------------------------------------------------------
// Persistent single-workgroup RNN scan.
// 16 waves (512 threads); wave w owns N-tile w (16 h-columns) for ALL 4
// M-tiles (batch rows).  Whh^T / Wih^T B-fragments resident in VGPRs
// (64 + 16 VGPRs, no duplication pressure); h double-buffered in LDS (f16).
// One workgroup barrier per timestep.  40 WMMAs per wave per step.
// ---------------------------------------------------------------------------
__global__ __launch_bounds__(512, 1)
void rnn_scan_kernel(const float* __restrict__ x,    // [B,T,DIN]
                     const float* __restrict__ h0,   // [1,H]
                     const float* __restrict__ Wih,  // [H,DIN]
                     const float* __restrict__ Whh,  // [H,H]
                     const float* __restrict__ bih,  // [H]
                     const float* __restrict__ bhh,  // [H]
                     float* __restrict__ hiddens)    // [B,T,H]
{
    __shared__ __align__(16) _Float16 hbuf[2][B_][HS];
    __shared__ __align__(16) _Float16 xbuf[2][B_][XS];

    const int tid  = threadIdx.x;
    const int wave = tid >> 5;      // 0..15  == N-tile index
    const int lane = tid & 31;
    const int g    = lane >> 4;
    const int ln   = lane & 15;

    const int n = wave * 16 + ln;   // output h-column owned by this lane

    // ---- one-time: load B fragments (Whh^T, Wih^T) into registers --------
    v16h whh[8], wih[2];
#pragma unroll
    for (int kk = 0; kk < 8; ++kk) {
#pragma unroll
        for (int j = 0; j < 16; ++j) {
            const int k = kk * 32 + kB(j, g);
            whh[kk][j] = (_Float16)Whh[n * H_ + k];   // B[k][n] = Whh[n][k]
        }
    }
#pragma unroll
    for (int kk = 0; kk < 2; ++kk) {
#pragma unroll
        for (int j = 0; j < 16; ++j) {
            const int k = kk * 32 + kB(j, g);
            wih[kk][j] = (_Float16)Wih[n * DIN + k];
        }
    }
    const float bias = bih[n] + bhh[n];

    // ---- init h(0) from h0 (broadcast over batch) ------------------------
    for (int i = tid; i < B_ * H_; i += 512) {
        const int b = i >> 8, hh = i & 255;
        hbuf[0][b][hh] = (_Float16)h0[hh];
    }

    // cooperative-x-load geometry (constant over t): 8 floats per thread
    const int xb = tid >> 3;          // batch row 0..63
    const int xd = (tid & 7) * 8;     // 8 floats each

    __syncthreads();

    for (int t = 0; t < T_; ++t) {
        const int cur = t & 1, nxt = cur ^ 1;

        // stage x_t -> LDS (f32 -> f16), prefetch x_{t+1}
        {
            const float* px = x + ((size_t)xb * T_ + t) * DIN + xd;
            v4f a0 = *(const v4f*)px;
            v4f a1 = *(const v4f*)(px + 4);
            if (t + 1 < T_) __builtin_prefetch(px + DIN, 0, 3);
            xbuf[cur][xb][xd + 0] = (_Float16)a0.x;
            xbuf[cur][xb][xd + 1] = (_Float16)a0.y;
            xbuf[cur][xb][xd + 2] = (_Float16)a0.z;
            xbuf[cur][xb][xd + 3] = (_Float16)a0.w;
            xbuf[cur][xb][xd + 4] = (_Float16)a1.x;
            xbuf[cur][xb][xd + 5] = (_Float16)a1.y;
            xbuf[cur][xb][xd + 6] = (_Float16)a1.z;
            xbuf[cur][xb][xd + 7] = (_Float16)a1.w;
        }
        __syncthreads();   // x_t ready; h writes of step t-1 complete

        v8f c[4];
#pragma unroll
        for (int m = 0; m < 4; ++m)
#pragma unroll
            for (int v = 0; v < 8; ++v) c[m][v] = bias;

        // input projection: pre += x_t @ Wih^T  (K = 64)
#pragma unroll
        for (int kk = 0; kk < 2; ++kk) {
#pragma unroll
            for (int m = 0; m < 4; ++m) {
                union { v16h v; v8h h[2]; } a;
                const int mrow = m * 16 + ln;           // A row, M = ln
                a.h[0] = *(const v8h*)&xbuf[cur][mrow][kk * 32 + 8 * g];
                a.h[1] = *(const v8h*)&xbuf[cur][mrow][kk * 32 + 16 + 8 * g];
                c[m] = __builtin_amdgcn_wmma_f32_16x16x32_f16(
                           false, a.v, false, wih[kk], (short)0, c[m],
                           false, false);
            }
        }
        // recurrent projection: += h @ Whh^T  (K = 256)
#pragma unroll
        for (int kk = 0; kk < 8; ++kk) {
#pragma unroll
            for (int m = 0; m < 4; ++m) {
                union { v16h v; v8h h[2]; } a;
                const int mrow = m * 16 + ln;
                a.h[0] = *(const v8h*)&hbuf[cur][mrow][kk * 32 + 8 * g];
                a.h[1] = *(const v8h*)&hbuf[cur][mrow][kk * 32 + 16 + 8 * g];
                c[m] = __builtin_amdgcn_wmma_f32_16x16x32_f16(
                           false, a.v, false, whh[kk], (short)0, c[m],
                           false, false);
            }
        }

        // tanh, publish h_{t+1} to LDS[nxt], stream hiddens to HBM (f32)
#pragma unroll
        for (int m = 0; m < 4; ++m) {
#pragma unroll
            for (int v = 0; v < 8; ++v) {
                const int brow = m * 16 + 8 * g + v;     // C/D: M = 8g+v
                const float tv = tanhf(c[m][v]);
                hbuf[nxt][brow][n] = (_Float16)tv;
                hiddens[((size_t)brow * T_ + t) * H_ + n] = tv;
            }
        }
    }
}

// ---------------------------------------------------------------------------
// Output projection: out[r,:] = hiddens[r,:] @ Wout^T + bout
// M = B*T = 131072, K = 256, N = 64.  One wave -> fixed nTile, 8 M-tiles;
// Wout B-fragments held in registers, hiddens streamed (f32 -> f16).
// ---------------------------------------------------------------------------
__global__ __launch_bounds__(256, 1)
void rnn_out_kernel(const float* __restrict__ hiddens, // [B*T, H]
                    const float* __restrict__ Wout,    // [DOUT, H]
                    const float* __restrict__ bout,    // [DOUT]
                    float* __restrict__ out)           // [B*T, DOUT]
{
    const int tid  = threadIdx.x;
    const int wid  = tid >> 5;
    const int lane = tid & 31;
    const int g    = lane >> 4;
    const int ln   = lane & 15;

    const int W     = blockIdx.x * 8 + wid;  // 0..4095
    const int nTile = W & 3;
    const int mBase = (W >> 2) * 8;          // 8 consecutive M-tiles per wave
    const int n     = nTile * 16 + ln;

    v16h bf[8];
#pragma unroll
    for (int kk = 0; kk < 8; ++kk)
#pragma unroll
        for (int j = 0; j < 16; ++j)
            bf[kk][j] = (_Float16)Wout[n * H_ + kk * 32 + kB(j, g)];
    const float bias = bout[n];

    for (int mt = 0; mt < 8; ++mt) {
        const size_t row = (size_t)(mBase + mt) * 16 + ln;   // A row, M = ln
        v8f c = {};
#pragma unroll
        for (int kk = 0; kk < 8; ++kk) {
            const float* p = hiddens + row * H_ + kk * 32 + 8 * g;
            v4f x0 = *(const v4f*)(p + 0);
            v4f x1 = *(const v4f*)(p + 4);
            v4f y0 = *(const v4f*)(p + 16);
            v4f y1 = *(const v4f*)(p + 20);
            float tmp[16];
            *(v4f*)&tmp[0]  = x0;  *(v4f*)&tmp[4]  = x1;
            *(v4f*)&tmp[8]  = y0;  *(v4f*)&tmp[12] = y1;
            v16h a;
#pragma unroll
            for (int j = 0; j < 16; ++j) a[j] = (_Float16)tmp[j];
            c = __builtin_amdgcn_wmma_f32_16x16x32_f16(false, a, false, bf[kk],
                                                       (short)0, c, false, false);
        }
#pragma unroll
        for (int v = 0; v < 8; ++v) {
            const size_t orow = (size_t)(mBase + mt) * 16 + 8 * g + v;
            out[orow * DOUT + n] = c[v] + bias;
        }
    }
}

extern "C" void kernel_launch(void* const* d_in, const int* in_sizes, int n_in,
                              void* d_out, int out_size, void* d_ws, size_t ws_size,
                              hipStream_t stream) {
    const float* x    = (const float*)d_in[0];
    const float* h0   = (const float*)d_in[1];
    const float* Wih  = (const float*)d_in[2];
    const float* Whh  = (const float*)d_in[3];
    const float* bih  = (const float*)d_in[4];
    const float* bhh  = (const float*)d_in[5];
    const float* Wout = (const float*)d_in[6];
    const float* bout = (const float*)d_in[7];

    float* outputs = (float*)d_out;                        // [B,T,DOUT]
    float* hiddens = outputs + (size_t)B_ * T_ * DOUT;     // [B,T,H]

    rnn_scan_kernel<<<dim3(1), dim3(512), 0, stream>>>(x, h0, Wih, Whh,
                                                       bih, bhh, hiddens);
    rnn_out_kernel<<<dim3(512), dim3(256), 0, stream>>>(hiddens, Wout, bout,
                                                        outputs);
}
